// GATAdapter_30777735643946
// MI455X (gfx1250) — compile-verified, matrix operand order checked by reading, not run on hardware
//
#include <hip/hip_runtime.h>

// ---------------- problem constants (from reference) ----------------
constexpr int kN    = 4096;
constexpr int kE    = 16384;
constexpr int kCLIP = 512;
constexpr int kHID  = 1024;
constexpr int kH    = 4;
constexpr int kOUT2 = 2048;   // P*MD
constexpr int kB    = 32;
constexpr int kNE   = kE + kN;      // edges + self loops
constexpr int kHC1  = kH * kHID;    // 4096
constexpr int kHC2  = kH * kOUT2;   // 8192

typedef __attribute__((ext_vector_type(16))) _Float16 v16h;
typedef __attribute__((ext_vector_type(8)))  float    v8f;

// ---------------- helpers ----------------
__device__ __forceinline__ unsigned f32_ord(float f) {
  unsigned u = __float_as_uint(f);
  return (u >> 31) ? ~u : (u ^ 0x80000000u);
}
__device__ __forceinline__ float ord_f32(unsigned o) {
  unsigned u = (o >> 31) ? (o ^ 0x80000000u) : ~o;
  return __uint_as_float(u);
}

// gfx1250 async copy: 16 bytes global -> LDS, tracked by ASYNCcnt.
__device__ __forceinline__ void async_cp16(unsigned ldsByte, const _Float16* g) {
  asm volatile("global_load_async_to_lds_b128 %0, %1, off"
               :: "v"(ldsByte), "v"((unsigned long long)(uintptr_t)g)
               : "memory");
}
__device__ __forceinline__ void wait_async0() {
  asm volatile("s_wait_asynccnt 0x0" ::: "memory");
}

__global__ void k_fill_f32(float* p, float v, size_t n) {
  size_t i = (size_t)blockIdx.x * blockDim.x + threadIdx.x;
  if (i < n) p[i] = v;
}
__global__ void k_fill_u32(unsigned* p, unsigned v, size_t n) {
  size_t i = (size_t)blockIdx.x * blockDim.x + threadIdx.x;
  if (i < n) p[i] = v;
}
// f32 -> f16 (same layout)
__global__ void k_cvt_h(const float* __restrict__ s, _Float16* __restrict__ d, size_t n) {
  size_t i = (size_t)blockIdx.x * blockDim.x + threadIdx.x;
  if (i < n) d[i] = (_Float16)s[i];
}
// f32[R][C] -> f16[C][R]  (transpose for B operand: Bt[n][k] = B[k][n])
__global__ void k_cvt_hT(const float* __restrict__ s, _Float16* __restrict__ d, int R, int C) {
  size_t i = (size_t)blockIdx.x * blockDim.x + threadIdx.x;
  if (i >= (size_t)R * C) return;
  int r = (int)(i / C), c = (int)(i % C);
  d[(size_t)c * R + r] = (_Float16)s[i];
}

// ---------------- WMMA GEMM (async-LDS, double buffered) -------------------
// C[M,Nn] = A[M,K] @ Bt[Nn,K]^T  (+bias, +prelu).  A,Bt are f16, K-contiguous.
// Block: 256 threads (8 waves). Tile 128(M) x 64(N), K-step 32.
// Waves: 4 along M x 2 along N; each wave -> 32x32 = 2x2 WMMA 16x16 tiles.
// Tiles staged with global_load_async_to_lds_b128; s_wait_asynccnt pipelining.
#define TM 128
#define TN 64
#define TK 32
#define LDK 40   // halves per LDS row (80B: 16B-aligned frags, 20-bank stride)

__global__ void __launch_bounds__(256)
k_gemm_wmma(const _Float16* __restrict__ A, const _Float16* __restrict__ Bt,
            float* __restrict__ C, int M, int K, int Nn,
            const float* __restrict__ bias, const float* __restrict__ preluA)
{
  __shared__ alignas(16) _Float16 Ah[2][TM * LDK];
  __shared__ alignas(16) _Float16 Bh[2][TN * LDK];

  const int tid   = threadIdx.x;
  const int lane  = tid & 31;
  const int wave  = tid >> 5;
  const int waveM = wave & 3;     // 0..3
  const int waveN = wave >> 2;    // 0..1
  const int blockM = blockIdx.y * TM;
  const int blockN = blockIdx.x * TN;

  v8f acc[2][2] = {};

  const int frow  = lane & 15;               // M (A frag) / N (B frag) index
  const int kHalf = (lane < 16) ? 0 : 8;     // per-lane K phase (wave32 layout)

  auto stageA = [&](int buf, int k0) {
    const _Float16* gA = A + (size_t)blockM * K + k0;
    #pragma unroll
    for (int t = 0; t < 2; ++t) {            // 512 16B chunks / 256 threads
      int i = tid + t * 256;
      int r = i >> 2, c8 = (i & 3) << 3;
      async_cp16((unsigned)(uintptr_t)&Ah[buf][r * LDK + c8],
                 gA + (size_t)r * K + c8);
    }
  };
  auto stageB = [&](int buf, int k0) {
    const _Float16* gB = Bt + (size_t)blockN * K + k0;
    int r = tid >> 2, c8 = (tid & 3) << 3;   // 256 chunks
    async_cp16((unsigned)(uintptr_t)&Bh[buf][r * LDK + c8],
               gB + (size_t)r * K + c8);
  };

  stageA(0, 0);
  stageB(0, 0);

  const int nk = K / TK;
  for (int kt = 0; kt < nk; ++kt) {
    wait_async0();        // our copies into buf[kt&1] landed in LDS
    __syncthreads();      // everyone's copies landed; prev compute done
    if (kt + 1 < nk) {    // overlap: stage next tile while computing this one
      stageA((kt + 1) & 1, (kt + 1) * TK);
      stageB((kt + 1) & 1, (kt + 1) * TK);
    }
    const int buf = kt & 1;

    #pragma unroll
    for (int mi = 0; mi < 2; ++mi) {
      v16h afrag;
      const _Float16* ap = &Ah[buf][(waveM * 32 + mi * 16 + frow) * LDK];
      #pragma unroll
      for (int j = 0; j < 8; ++j) {          // two 16B ds_load_b128 runs
        afrag[j]     = ap[kHalf + j];
        afrag[j + 8] = ap[kHalf + 16 + j];
      }
      #pragma unroll
      for (int ni = 0; ni < 2; ++ni) {
        v16h bfrag;
        const _Float16* bp = &Bh[buf][(waveN * 32 + ni * 16 + frow) * LDK];
        #pragma unroll
        for (int j = 0; j < 8; ++j) {        // contiguous: vectorizes too
          bfrag[j]     = bp[kHalf + j];
          bfrag[j + 8] = bp[kHalf + 16 + j];
        }
        acc[mi][ni] = __builtin_amdgcn_wmma_f32_16x16x32_f16(
            false, afrag, false, bfrag, (short)0, acc[mi][ni], false, false);
      }
    }
    // no trailing barrier: next iteration's wait+barrier orders the overwrite
  }

  const float alpha = preluA ? preluA[0] : 0.f;
  const int mBase = (lane < 16) ? 0 : 8;
  #pragma unroll
  for (int mi = 0; mi < 2; ++mi)
    #pragma unroll
    for (int ni = 0; ni < 2; ++ni) {
      const int col = blockN + waveN * 32 + ni * 16 + (lane & 15);
      const float bv = bias ? bias[col] : 0.f;
      #pragma unroll
      for (int r = 0; r < 8; ++r) {
        const int row = blockM + waveM * 32 + mi * 16 + mBase + r;
        float v = acc[mi][ni][r] + bv;
        if (preluA) v = (v >= 0.f) ? v : alpha * v;
        C[(size_t)row * Nn + col] = v;
      }
    }
}

// ---------------- self-loop edge_attr = mean of incoming -------------------
__global__ void k_loop_accum(const int* __restrict__ ei,
                             const float* __restrict__ ea,
                             float* __restrict__ sums, float* __restrict__ cnt) {
  int e = blockIdx.x;
  int d = ei[kE + e];
  for (int f = threadIdx.x; f < kCLIP; f += blockDim.x)
    atomicAdd(&sums[(size_t)d * kCLIP + f], ea[(size_t)e * kCLIP + f]);
  if (threadIdx.x == 0) atomicAdd(&cnt[d], 1.f);
}
__global__ void k_loop_div(float* __restrict__ sums, const float* __restrict__ cnt) {
  size_t i = (size_t)blockIdx.x * blockDim.x + threadIdx.x;
  if (i >= (size_t)kN * kCLIP) return;
  sums[i] /= fmaxf(cnt[i / kCLIP], 1.f);
}

// ---------------- fold We with ae: ve[k,h] = sum_c We[k,h*C+c]*ae[h,c] -----
__global__ void k_fold_ae(const float* __restrict__ We, const float* __restrict__ aev,
                          float* __restrict__ ve, int C) {
  int i = blockIdx.x * blockDim.x + threadIdx.x;
  if (i >= kCLIP * kH) return;
  int k = i / kH, h = i % kH;
  float s = 0.f;
  const float* wr = We + (size_t)k * (kH * C) + (size_t)h * C;
  const float* ar = aev + (size_t)h * C;
  for (int c = 0; c < C; ++c) s += wr[c] * ar[c];
  ve[k * kH + h] = s;
}

// ---------------- a_e[e',h] = row(e') . ve[:,h]  (self loops use loop_attr)
__global__ void k_edge_ae(const float* __restrict__ ea, const float* __restrict__ loop_attr,
                          const float* __restrict__ ve, float* __restrict__ aeb) {
  int i = blockIdx.x * blockDim.x + threadIdx.x;
  if (i >= kNE * kH) return;
  int e = i / kH, h = i % kH;
  const float* row = (e < kE) ? ea + (size_t)e * kCLIP
                              : loop_attr + (size_t)(e - kE) * kCLIP;
  float s = 0.f;
  for (int k = 0; k < kCLIP; ++k) s += row[k] * ve[k * kH + h];
  aeb[i] = s;
}

// ---------------- node attention terms a_s,a_d from H = x@W ---------------
__global__ void k_node_att(const float* __restrict__ Hb,
                           const float* __restrict__ asrc, const float* __restrict__ adst,
                           float* __restrict__ as_, float* __restrict__ ad_, int C) {
  int i = blockIdx.x * blockDim.x + threadIdx.x;
  if (i >= kN * kH) return;
  int n = i / kH, h = i % kH;
  const float* hp = Hb + (size_t)n * (kH * C) + (size_t)h * C;
  const float* s1p = asrc + (size_t)h * C;
  const float* s2p = adst + (size_t)h * C;
  float s1 = 0.f, s2 = 0.f;
  for (int c = 0; c < C; ++c) { float v = hp[c]; s1 += v * s1p[c]; s2 += v * s2p[c]; }
  as_[i] = s1; ad_[i] = s2;
}

// ---------------- edge softmax (segment over dst, per head) ----------------
__global__ void k_attn_pass1(const int* __restrict__ ei,
                             const float* __restrict__ as_, const float* __restrict__ ad_,
                             const float* __restrict__ aeb,
                             float* __restrict__ alphab, unsigned* __restrict__ amax) {
  int i = blockIdx.x * blockDim.x + threadIdx.x;
  if (i >= kNE * kH) return;
  int e = i / kH, h = i % kH;
  int s, d;
  if (e < kE) { s = ei[e]; d = ei[kE + e]; } else { s = d = e - kE; }
  float a = as_[s * kH + h] + ad_[d * kH + h] + aeb[i];
  a = (a >= 0.f) ? a : 0.2f * a;   // leaky_relu 0.2
  alphab[i] = a;
  atomicMax(&amax[d * kH + h], f32_ord(a));
}
__global__ void k_attn_pass2(const int* __restrict__ ei,
                             float* __restrict__ alphab,
                             const unsigned* __restrict__ amax,
                             float* __restrict__ denom) {
  int i = blockIdx.x * blockDim.x + threadIdx.x;
  if (i >= kNE * kH) return;
  int e = i / kH, h = i % kH;
  int d = (e < kE) ? ei[kE + e] : (e - kE);
  float ex = __expf(alphab[i] - ord_f32(amax[d * kH + h]));
  alphab[i] = ex;
  atomicAdd(&denom[d * kH + h], ex);
}

// ---------------- weighted message scatter: acc[dst] += h[src]*w ----------
__global__ void k_scatter(const int* __restrict__ ei, const float* __restrict__ Hb,
                          const float* __restrict__ exb, const float* __restrict__ denom,
                          float* __restrict__ acc, int C, int cShift) {
  __shared__ float w[kH];
  int e = blockIdx.x;
  int s, d;
  if (e < kE) { s = ei[e]; d = ei[kE + e]; } else { s = d = e - kE; }
  if (threadIdx.x < kH)
    w[threadIdx.x] = exb[e * kH + threadIdx.x] /
                     fmaxf(denom[d * kH + threadIdx.x], 1e-16f);
  __syncthreads();
  const int HC = kH * C;
  for (int idx = threadIdx.x; idx < HC; idx += blockDim.x) {
    int h = idx >> cShift;
    atomicAdd(&acc[(size_t)d * HC + idx], Hb[(size_t)s * HC + idx] * w[h]);
  }
}

// ---------------- head-mean + bias + prelu --------------------------------
__global__ void k_finalize(const float* __restrict__ acc, const float* __restrict__ bias,
                           const float* __restrict__ pA, float* __restrict__ out, int C) {
  size_t i = (size_t)blockIdx.x * blockDim.x + threadIdx.x;
  if (i >= (size_t)kN * C) return;
  size_t n = i / C; int c = (int)(i % C);
  const float* ap = acc + n * (size_t)(kH * C) + c;
  float s = 0.f;
  for (int h = 0; h < kH; ++h) s += ap[(size_t)h * C];
  s = s * (1.f / kH) + bias[c];
  float a = pA[0];
  out[i] = (s >= 0.f) ? s : a * s;
}

// ---------------- pooling: gate3 = gate2 @ gW3 + gb3, segment softmax -----
__global__ void k_gate3(const float* __restrict__ g2, const float* __restrict__ gW3,
                        const float* __restrict__ gb3, const int* __restrict__ batch,
                        float* __restrict__ gate3, unsigned* __restrict__ gmax) {
  int n = blockIdx.x * blockDim.x + threadIdx.x;
  if (n >= kN) return;
  float s = gb3[0];
  const float* gp = g2 + (size_t)n * kHID;
  for (int c = 0; c < kHID; ++c) s += gp[c] * gW3[c];
  gate3[n] = s;
  atomicMax(&gmax[batch[n]], f32_ord(s));
}
__global__ void k_pool_pass2(const float* __restrict__ gate3, const int* __restrict__ batch,
                             const unsigned* __restrict__ gmax,
                             float* __restrict__ exn, float* __restrict__ gden) {
  int n = blockIdx.x * blockDim.x + threadIdx.x;
  if (n >= kN) return;
  int b = batch[n];
  float ex = __expf(gate3[n] - ord_f32(gmax[b]));
  exn[n] = ex;
  atomicAdd(&gden[b], ex);
}
__global__ void k_pool_scatter(const float* __restrict__ out2, const int* __restrict__ batch,
                               const float* __restrict__ exn, const float* __restrict__ gden,
                               float* __restrict__ pooled) {
  int n = blockIdx.x;
  int b = batch[n];
  float w = exn[n] / fmaxf(gden[b], 1e-16f);
  for (int f = threadIdx.x; f < kOUT2; f += blockDim.x)
    atomicAdd(&pooled[(size_t)b * kOUT2 + f], out2[(size_t)n * kOUT2 + f] * w);
}

// ---------------- orchestration -------------------------------------------
extern "C" void kernel_launch(void* const* d_in, const int* in_sizes, int n_in,
                              void* d_out, int out_size, void* d_ws, size_t ws_size,
                              hipStream_t stream) {
  (void)in_sizes; (void)n_in; (void)out_size; (void)ws_size;
  const float* x     = (const float*)d_in[0];
  const int*   ei    = (const int*)d_in[1];
  const float* ea    = (const float*)d_in[2];
  const int*   batch = (const int*)d_in[3];
  const float* W1    = (const float*)d_in[4];
  const float* asrc1 = (const float*)d_in[5];
  const float* adst1 = (const float*)d_in[6];
  const float* We1   = (const float*)d_in[7];
  const float* ae1   = (const float*)d_in[8];
  const float* b1    = (const float*)d_in[9];
  const float* p1    = (const float*)d_in[10];
  const float* W2    = (const float*)d_in[11];
  const float* asrc2 = (const float*)d_in[12];
  const float* adst2 = (const float*)d_in[13];
  const float* We2   = (const float*)d_in[14];
  const float* ae2   = (const float*)d_in[15];
  const float* b2    = (const float*)d_in[16];
  const float* p2    = (const float*)d_in[17];
  const float* gW1   = (const float*)d_in[18];
  const float* gb1   = (const float*)d_in[19];
  const float* gp1   = (const float*)d_in[20];
  const float* gW2   = (const float*)d_in[21];
  const float* gb2   = (const float*)d_in[22];
  const float* gp2   = (const float*)d_in[23];
  const float* gW3   = (const float*)d_in[24];
  const float* gb3   = (const float*)d_in[25];

  float* wsf = (float*)d_ws;
  size_t off = 0;
  auto alloc  = [&](size_t nf) { float* p = wsf + off; off += (nf + 63) & ~(size_t)63; return p; };
  auto allocH = [&](size_t nh) { return (_Float16*)alloc((nh + 1) / 2); };

  float*    loop_attr = alloc((size_t)kN * kCLIP);  // sums, then /= cnt in place
  float*    cnt   = alloc(kN);
  float*    ve1   = alloc(kCLIP * kH);
  float*    ve2   = alloc(kCLIP * kH);
  float*    ae1b  = alloc((size_t)kNE * kH);
  float*    ae2b  = alloc((size_t)kNE * kH);
  float*    as_b  = alloc(kN * kH);
  float*    ad_b  = alloc(kN * kH);
  float*    alphab= alloc((size_t)kNE * kH);
  unsigned* amax  = (unsigned*)alloc(kN * kH);
  float*    denom = alloc(kN * kH);
  float*    H1    = alloc((size_t)kN * kHC1);   // 64MB
  float*    acc1  = alloc((size_t)kN * kHC1);   // 64MB, contiguous after H1
  float*    H2    = H1;                         // layer2 GEMM reuses H1+acc1 (128MB)
  float*    out1  = alloc((size_t)kN * kHID);
  float*    acc2  = alloc((size_t)kN * kHC2);   // 128MB
  float*    out2  = alloc((size_t)kN * kOUT2);
  float*    gate1 = alloc((size_t)kN * kHID);
  float*    gate2 = alloc((size_t)kN * kHID);
  float*    gate3 = alloc(kN);
  float*    exn   = alloc(kN);
  unsigned* gmax  = (unsigned*)alloc(kB);
  float*    gden  = alloc(kB);
  // f16 operands for the WMMA GEMMs (A row-major, B pre-transposed to N x K)
  _Float16* x16   = allocH((size_t)kN * kCLIP);
  _Float16* W1t   = allocH((size_t)kHC1 * kCLIP);
  _Float16* o1_16 = allocH((size_t)kN * kHID);
  _Float16* W2t   = allocH((size_t)kHC2 * kHID);
  _Float16* o2_16 = allocH((size_t)kN * kOUT2);
  _Float16* gW1t  = allocH((size_t)kHID * kOUT2);
  _Float16* g1_16 = allocH((size_t)kN * kHID);
  _Float16* gW2t  = allocH((size_t)kHID * kHID);

  auto fillf = [&](float* p, float v, size_t n) {
    k_fill_f32<<<(unsigned)((n + 255) / 256), 256, 0, stream>>>(p, v, n);
  };
  auto fillu = [&](unsigned* p, unsigned v, size_t n) {
    k_fill_u32<<<(unsigned)((n + 255) / 256), 256, 0, stream>>>(p, v, n);
  };
  auto cvt = [&](const float* s, _Float16* d, size_t n) {
    k_cvt_h<<<(unsigned)((n + 255) / 256), 256, 0, stream>>>(s, d, n);
  };
  auto cvtT = [&](const float* s, _Float16* d, int R, int C) {
    k_cvt_hT<<<(unsigned)(((size_t)R * C + 255) / 256), 256, 0, stream>>>(s, d, R, C);
  };

  // ---- output + self-loop attr ----
  fillf((float*)d_out, 0.f, (size_t)kB * kOUT2);
  fillf(loop_attr, 0.f, (size_t)kN * kCLIP);
  fillf(cnt, 0.f, kN);
  k_loop_accum<<<kE, 256, 0, stream>>>(ei, ea, loop_attr, cnt);
  k_loop_div<<<(unsigned)(((size_t)kN * kCLIP + 255) / 256), 256, 0, stream>>>(loop_attr, cnt);

  // ---- fold edge-attention vectors; a_e for both layers ----
  k_fold_ae<<<(kCLIP * kH + 255) / 256, 256, 0, stream>>>(We1, ae1, ve1, kHID);
  k_fold_ae<<<(kCLIP * kH + 255) / 256, 256, 0, stream>>>(We2, ae2, ve2, kOUT2);
  k_edge_ae<<<(kNE * kH + 255) / 256, 256, 0, stream>>>(ea, loop_attr, ve1, ae1b);
  k_edge_ae<<<(kNE * kH + 255) / 256, 256, 0, stream>>>(ea, loop_attr, ve2, ae2b);

  // ---- one-time f16 conversions of inputs/weights ----
  cvt(x, x16, (size_t)kN * kCLIP);
  cvtT(W1, W1t, kCLIP, kHC1);
  cvtT(W2, W2t, kHID, kHC2);
  cvtT(gW1, gW1t, kOUT2, kHID);
  cvtT(gW2, gW2t, kHID, kHID);

  // ================= GAT layer 1 =================
  {
    dim3 grid(kHC1 / TN, kN / TM);
    k_gemm_wmma<<<grid, 256, 0, stream>>>(x16, W1t, H1, kN, kCLIP, kHC1, nullptr, nullptr);
  }
  k_node_att<<<(kN * kH + 255) / 256, 256, 0, stream>>>(H1, asrc1, adst1, as_b, ad_b, kHID);
  fillu(amax, 0u, kN * kH);
  fillf(denom, 0.f, kN * kH);
  k_attn_pass1<<<(kNE * kH + 255) / 256, 256, 0, stream>>>(ei, as_b, ad_b, ae1b, alphab, amax);
  k_attn_pass2<<<(kNE * kH + 255) / 256, 256, 0, stream>>>(ei, alphab, amax, denom);
  fillf(acc1, 0.f, (size_t)kN * kHC1);
  k_scatter<<<kNE, 256, 0, stream>>>(ei, H1, alphab, denom, acc1, kHID, 10);
  k_finalize<<<(unsigned)(((size_t)kN * kHID + 255) / 256), 256, 0, stream>>>(acc1, b1, p1, out1, kHID);

  // ================= GAT layer 2 =================
  cvt(out1, o1_16, (size_t)kN * kHID);
  {
    dim3 grid(kHC2 / TN, kN / TM);
    k_gemm_wmma<<<grid, 256, 0, stream>>>(o1_16, W2t, H2, kN, kHID, kHC2, nullptr, nullptr);
  }
  k_node_att<<<(kN * kH + 255) / 256, 256, 0, stream>>>(H2, asrc2, adst2, as_b, ad_b, kOUT2);
  fillu(amax, 0u, kN * kH);
  fillf(denom, 0.f, kN * kH);
  k_attn_pass1<<<(kNE * kH + 255) / 256, 256, 0, stream>>>(ei, as_b, ad_b, ae2b, alphab, amax);
  k_attn_pass2<<<(kNE * kH + 255) / 256, 256, 0, stream>>>(ei, alphab, amax, denom);
  fillf(acc2, 0.f, (size_t)kN * kHC2);
  k_scatter<<<kNE, 256, 0, stream>>>(ei, H2, alphab, denom, acc2, kOUT2, 11);
  k_finalize<<<(unsigned)(((size_t)kN * kOUT2 + 255) / 256), 256, 0, stream>>>(acc2, b2, p2, out2, kOUT2);

  // ================= attentional pooling =================
  cvt(out2, o2_16, (size_t)kN * kOUT2);
  {
    dim3 grid(kHID / TN, kN / TM);
    k_gemm_wmma<<<grid, 256, 0, stream>>>(o2_16, gW1t, gate1, kN, kOUT2, kHID, gb1, gp1);
  }
  cvt(gate1, g1_16, (size_t)kN * kHID);
  {
    dim3 grid(kHID / TN, kN / TM);
    k_gemm_wmma<<<grid, 256, 0, stream>>>(g1_16, gW2t, gate2, kN, kHID, kHID, gb2, gp2);
  }
  fillu(gmax, 0u, kB);
  fillf(gden, 0.f, kB);
  k_gate3<<<(kN + 255) / 256, 256, 0, stream>>>(gate2, gW3, gb3, batch, gate3, gmax);
  k_pool_pass2<<<(kN + 255) / 256, 256, 0, stream>>>(gate3, batch, gmax, exn, gden);
  k_pool_scatter<<<kN, 256, 0, stream>>>(out2, batch, exn, gden, (float*)d_out);
}
